// QuantizedLinear_16028817948933
// MI455X (gfx1250) — compile-verified, matrix operand order checked by reading, not run on hardware
//
#include <hip/hip_runtime.h>

// ---------------------------------------------------------------------------
// shift_matmul on gfx1250: out[512,16384] = x[512,4096] . W^T + bias
// W = q_weight(int32 in {-4,-2,-1,0,1,2,4}) * fp16 scale per 64-wide K-block.
// Strategy: read the 256MB int32 weight once per 128-row M tile (4 passes), dequantize on the fly
// into fp16 (exact for power-of-two levels), stage x/W tiles through LDS and
// run v_wmma_f32_16x16x32_f16 with f32 accumulation.
// ---------------------------------------------------------------------------

typedef __attribute__((ext_vector_type(16))) _Float16 v16h;
typedef __attribute__((ext_vector_type(8)))  _Float16 v8h;
typedef __attribute__((ext_vector_type(8)))  float    v8f;
typedef __attribute__((ext_vector_type(4)))  float    f32x4;
typedef __attribute__((ext_vector_type(4)))  int      i32x4;

#define IN_F   4096
#define OUT_F  16384
#define SEQ    512
#define NBLK   64            // IN_F / 64 scale blocks per output row
#define BM     128           // M tile per workgroup
#define BN     128           // N tile per workgroup
#define BK     64            // K stage (== one scale block)
#define LDS_K  72            // padded K stride in halves (144B rows, 16B aligned)
#define KSTAGES (IN_F / BK)  // 64

__global__ __launch_bounds__(256)
void shift_gemm_wmma(const float*    __restrict__ x,
                     const int*      __restrict__ qw,
                     const _Float16* __restrict__ scales,
                     const _Float16* __restrict__ bias,
                     float*          __restrict__ out)
{
    __shared__ _Float16 lx[BM * LDS_K];   // x tile, fp16, row-major [m][k]
    __shared__ _Float16 lw[BN * LDS_K];   // dequantized W tile, fp16, [n][k]

    const int tid  = threadIdx.x;
    const int lane = tid & 31;
    const int wav  = tid >> 5;            // 0..7
    const int wm   = wav >> 2;            // 0..1 : 64 M-rows per wave
    const int wn   = wav & 3;             // 0..3 : 32 N-cols per wave

    const int n_base = blockIdx.x * BN;
    const int m_base = blockIdx.y * BM;

    // Loader mapping: each thread fills half a row (32 elements) of each tile.
    const int lrow = tid >> 1;            // 0..127
    const int lk   = (tid & 1) * 32;      // 0 or 32

    const float*    xg = x  + (size_t)(m_base + lrow) * IN_F + lk;
    const int*      wg = qw + (size_t)(n_base + lrow) * IN_F + lk;
    const _Float16* sg = scales + (size_t)(n_base + lrow) * NBLK;

    v8f acc[4][2];
    #pragma unroll
    for (int mt = 0; mt < 4; ++mt)
        #pragma unroll
        for (int nt = 0; nt < 2; ++nt)
            acc[mt][nt] = (v8f){0.f, 0.f, 0.f, 0.f, 0.f, 0.f, 0.f, 0.f};

    // WMMA fp16 A/B fragment K-mapping (wave32):
    //   lanes 0-15 : halves 0..7 -> K 0..7,  halves 8..15 -> K 16..23
    //   lanes 16-31: halves 0..7 -> K 8..15, halves 8..15 -> K 24..31
    const int row16 = lane & 15;
    const int kbase = (lane < 16) ? 0 : 8;

    for (int kt = 0; kt < KSTAGES; ++kt) {
        const int k0 = kt * BK;

        // ---- global -> registers, dequantize to fp16 -------------------
        const float s = (float)sg[kt];    // one scale block per BK stage
        _Float16 cx[32];
        _Float16 cw[32];
        #pragma unroll
        for (int j = 0; j < 8; ++j) {
            f32x4 xf = ((const f32x4*)(xg + k0))[j];
            i32x4 qv = ((const i32x4*)(wg + k0))[j];
            #pragma unroll
            for (int e = 0; e < 4; ++e) {
                cx[j * 4 + e] = (_Float16)xf[e];
                cw[j * 4 + e] = (_Float16)((float)qv[e] * s);
            }
        }

        // Prefetch next K stage while this one computes (global_prefetch_b8).
        if (kt + 1 < KSTAGES) {
            __builtin_prefetch(xg + k0 + BK, 0, 3);
            __builtin_prefetch(wg + k0 + BK, 0, 3);
        }

        __syncthreads();                  // prior iteration done reading LDS

        // ---- registers -> LDS (16B ds_store_b128, all aligned) ---------
        {
            v8h* dx = (v8h*)&lx[lrow * LDS_K + lk];
            v8h* dw = (v8h*)&lw[lrow * LDS_K + lk];
            const v8h* sx = (const v8h*)cx;
            const v8h* sw = (const v8h*)cw;
            #pragma unroll
            for (int j = 0; j < 4; ++j) { dx[j] = sx[j]; dw[j] = sw[j]; }
        }

        __syncthreads();                  // tile visible to all waves

        // ---- two 16x16x32 k-chunks per stage ---------------------------
        #pragma unroll
        for (int kc = 0; kc < 2; ++kc) {
            union Frag { v16h v; v8h h[2]; };
            Frag a[4], b[2];
            #pragma unroll
            for (int mt = 0; mt < 4; ++mt) {
                const _Float16* p =
                    &lx[(wm * 64 + mt * 16 + row16) * LDS_K + kc * 32 + kbase];
                a[mt].h[0] = *(const v8h*)p;
                a[mt].h[1] = *(const v8h*)(p + 16);
            }
            #pragma unroll
            for (int nt = 0; nt < 2; ++nt) {
                const _Float16* p =
                    &lw[(wn * 32 + nt * 16 + row16) * LDS_K + kc * 32 + kbase];
                b[nt].h[0] = *(const v8h*)p;
                b[nt].h[1] = *(const v8h*)(p + 16);
            }
            #pragma unroll
            for (int mt = 0; mt < 4; ++mt)
                #pragma unroll
                for (int nt = 0; nt < 2; ++nt)
                    acc[mt][nt] = __builtin_amdgcn_wmma_f32_16x16x32_f16(
                        /*neg_a=*/false, a[mt].v,
                        /*neg_b=*/false, b[nt].v,
                        /*c_mod=*/(short)0, acc[mt][nt],
                        /*reuse_a=*/false, /*reuse_b=*/false);
        }

        __syncthreads();                  // computes done before next fill
    }

    // ---- epilogue: +bias, store f32 ------------------------------------
    // C/D layout: VGPR v -> M = v (lanes 0-15) or v+8 (lanes 16-31); N = lane%16.
    #pragma unroll
    for (int nt = 0; nt < 2; ++nt) {
        const int n  = n_base + wn * 32 + nt * 16 + row16;
        const float bf = (float)bias[n];
        #pragma unroll
        for (int mt = 0; mt < 4; ++mt) {
            const int m0 = m_base + wm * 64 + mt * 16 + ((lane >> 4) << 3);
            #pragma unroll
            for (int v = 0; v < 8; ++v)
                out[(size_t)(m0 + v) * OUT_F + n] = acc[mt][nt][v] + bf;
        }
    }
}

extern "C" void kernel_launch(void* const* d_in, const int* in_sizes, int n_in,
                              void* d_out, int out_size, void* d_ws, size_t ws_size,
                              hipStream_t stream) {
    (void)in_sizes; (void)n_in; (void)out_size; (void)d_ws; (void)ws_size;
    const float*    x      = (const float*)d_in[0];     // [512, 4096] f32
    const int*      qw     = (const int*)d_in[1];       // [16384, 4096] i32
    const _Float16* scales = (const _Float16*)d_in[2];  // [16384, 64] f16
    const _Float16* bias   = (const _Float16*)d_in[3];  // [16384] f16
    float*          out    = (float*)d_out;             // [512, 16384] f32

    dim3 grid(OUT_F / BN, SEQ / BM);   // 128 x 4 = 512 workgroups
    dim3 block(256);                   // 8 wave32 per workgroup
    shift_gemm_wmma<<<grid, block, 0, stream>>>(x, qw, scales, bias, out);
}
